// EAGLE2Decoder_4440996184669
// MI455X (gfx1250) — compile-verified, matrix operand order checked by reading, not run on hardware
//
#include <hip/hip_runtime.h>
#include <math.h>

#define Bc   2
#define Tc   512
#define DMc  1024
#define DHc  4096
#define DDc  4
#define Vc   32000
#define Kbr  8
#define EPSc 1e-5f

typedef __attribute__((ext_vector_type(16))) __bf16 bf16x16;
typedef __attribute__((ext_vector_type(8)))  float  f32x8;
typedef __attribute__((ext_vector_type(4)))  int    v4i;

union Frag {
    unsigned int   u32[8];
    unsigned short u16[16];
    bf16x16        v;
};

__device__ __forceinline__ unsigned short f2bf(float f) {
    unsigned int u = __float_as_uint(f);
    u += 0x7FFFu + ((u >> 16) & 1u);            // round-to-nearest-even
    return (unsigned short)(u >> 16);
}

// ---------------------------------------------------------------------------
// gfx1250 async global->LDS copy (ASYNCcnt) with plain-copy fallback.
// Probe result: builtin expects (int4 addrspace(1)*, int4 addrspace(3)*, ...).
// ---------------------------------------------------------------------------
#if defined(__HIP_DEVICE_COMPILE__) && \
    __has_builtin(__builtin_amdgcn_global_load_async_to_lds_b128)
#define HAVE_ASYNC_LDS 1
#endif

__device__ __forceinline__ void copy16_g2l(const void* g, void* l) {
#if defined(HAVE_ASYNC_LDS)
    __builtin_amdgcn_global_load_async_to_lds_b128(
        (__attribute__((address_space(1))) v4i*)(unsigned long long)(size_t)g,
        (__attribute__((address_space(3))) v4i*)(unsigned)(size_t)l,
        0, 0);
#else
    *(uint4*)l = *(const uint4*)g;
#endif
}

__device__ __forceinline__ void async_copy_wait() {
#if defined(HAVE_ASYNC_LDS)
#if __has_builtin(__builtin_amdgcn_s_wait_asynccnt)
    __builtin_amdgcn_s_wait_asynccnt(0);
#else
    asm volatile("s_wait_asynccnt 0" ::: "memory");
#endif
#endif
}

// ---------------------------------------------------------------------------
// One-time weight transpose + f32->bf16 convert: Wt[n][k] = bf16(W[k][n])
// 32x32 tiles through LDS, coalesced both ways.
// ---------------------------------------------------------------------------
__global__ __launch_bounds__(256) void convert_transpose_kernel(
    const float* __restrict__ W, unsigned short* __restrict__ Wt, int K, int N)
{
    __shared__ unsigned short t[32][33];
    const int bk = blockIdx.y * 32, bn = blockIdx.x * 32;
    const int r  = threadIdx.x >> 3;          // 0..31
    const int c4 = (threadIdx.x & 7) * 4;     // 0,4,...,28

    const float4 v = *(const float4*)(W + (size_t)(bk + r) * N + bn + c4);
    t[c4 + 0][r] = f2bf(v.x);
    t[c4 + 1][r] = f2bf(v.y);
    t[c4 + 2][r] = f2bf(v.z);
    t[c4 + 3][r] = f2bf(v.w);
    __syncthreads();

    unsigned short* dst = Wt + (size_t)(bn + r) * K + bk + c4;
    uint2 o;
    o.x = (unsigned)t[r][c4 + 0] | ((unsigned)t[r][c4 + 1] << 16);
    o.y = (unsigned)t[r][c4 + 2] | ((unsigned)t[r][c4 + 3] << 16);
    *(uint2*)dst = o;
}

// ---------------------------------------------------------------------------
// Tiled bf16 WMMA GEMM: C = epilogue(A[MxK]bf16 @ Bt[NxK]bf16^T + bias)
// Block: 256 threads (8 waves). Tile 128x128, BK=32.
// Wave grid 4(M)x2(N); each wave owns 2x4 16x16 f32 accumulators.
// A and Bt tiles have identical [row][k] layout -> staging is pure 16B
// async copies; fragment gathers merge into ds_load_b128.
// EPI 0: exact GELU -> bf16 out       (U = gelu(S@W1+b1))
// EPI 1: bias       -> f32 + bf16 out (S2 = U@W2+b2)
// EPI 2: bias+mask  -> f32 out        (q  = S2@Wq+bq, zeroed where !mask)
// ---------------------------------------------------------------------------
template <int EPI>
__global__ __launch_bounds__(256) void gemm_bf16_wmma(
    const unsigned short* __restrict__ A,   // M x K, bf16 row-major
    const unsigned short* __restrict__ Bt,  // N x K, bf16 (weights pre-transposed)
    const float* __restrict__ bias, int M, int N, int K,
    unsigned short* __restrict__ outH, float* __restrict__ outF,
    const int* __restrict__ mask)
{
    __shared__ __align__(16) unsigned short Alds[128 * 32];   // [m][k]
    __shared__ __align__(16) unsigned short Blds[128 * 32];   // [n][k]

    const int tid  = threadIdx.x;
    const int lane = tid & 31;
    const int wid  = tid >> 5;
    const int wm   = (wid & 3) * 32;          // wave M offset in tile
    const int wn   = (wid >> 2) * 64;         // wave N offset in tile
    const int bm   = blockIdx.y * 128;
    const int bn   = blockIdx.x * 128;

    const f32x8 zero = {0.f, 0.f, 0.f, 0.f, 0.f, 0.f, 0.f, 0.f};
    f32x8 acc[2][4];
#pragma unroll
    for (int i = 0; i < 2; ++i)
#pragma unroll
        for (int j = 0; j < 4; ++j) acc[i][j] = zero;

    const int srow = tid >> 1, sseg = (tid & 1) * 16;   // 128 rows x 32 bf16

    const unsigned short* aG = A  + (size_t)(bm + srow) * K + sseg;
    const unsigned short* bG = Bt + (size_t)(bn + srow) * K + sseg;
    unsigned short* aL = Alds + srow * 32 + sseg;
    unsigned short* bL = Blds + srow * 32 + sseg;

    for (int k0 = 0; k0 < K; k0 += 32) {
        copy16_g2l(aG + k0,     aL);
        copy16_g2l(aG + k0 + 8, aL + 8);
        copy16_g2l(bG + k0,     bL);
        copy16_g2l(bG + k0 + 8, bL + 8);
#if !defined(HAVE_ASYNC_LDS)
        if (k0 + 32 < K) {  // emits global_prefetch_b8 for next tiles
            __builtin_prefetch(aG + k0 + 32, 0, 1);
            __builtin_prefetch(bG + k0 + 32, 0, 1);
        }
#endif
        async_copy_wait();
        __syncthreads();

        // per-lane fragment gather (16-bit A 16x32 VGPR layout)
        const int half = lane >> 4, ml = lane & 15;
        Frag af[2], bfr[4];
#pragma unroll
        for (int mt = 0; mt < 2; ++mt) {
            const unsigned short* row = Alds + (wm + mt * 16 + ml) * 32;
#pragma unroll
            for (int v = 0; v < 8; ++v) {
                int kk = (v < 4) ? (half * 8 + 2 * v) : (16 + half * 8 + 2 * (v - 4));
                af[mt].u32[v] = *(const unsigned int*)(row + kk);
            }
        }
#pragma unroll
        for (int nt = 0; nt < 4; ++nt) {
            const unsigned short* row = Blds + (wn + nt * 16 + ml) * 32;
#pragma unroll
            for (int v = 0; v < 8; ++v) {
                int kk = (v < 4) ? (half * 8 + 2 * v) : (16 + half * 8 + 2 * (v - 4));
                bfr[nt].u32[v] = *(const unsigned int*)(row + kk);
            }
        }

#pragma unroll
        for (int mt = 0; mt < 2; ++mt)
#pragma unroll
            for (int nt = 0; nt < 4; ++nt)
                acc[mt][nt] = __builtin_amdgcn_wmma_f32_16x16x32_bf16(
                    false, af[mt].v, false, bfr[nt].v, (short)0, acc[mt][nt],
                    false, false);

        __syncthreads();
    }

    // epilogue: C/D 16x16 f32 layout: VGPR r -> row r (lanes<16) / 8+r
    const int cm = (lane >> 4) * 8;
    const int cn = lane & 15;
#pragma unroll
    for (int mt = 0; mt < 2; ++mt) {
#pragma unroll
        for (int nt = 0; nt < 4; ++nt) {
            const int   gn = bn + wn + nt * 16 + cn;
            const float bv = bias[gn];
#pragma unroll
            for (int r = 0; r < 8; ++r) {
                const int gm  = bm + wm + mt * 16 + cm + r;
                float     val = acc[mt][nt][r] + bv;
                if (EPI == 0) {
                    val = 0.5f * val * (1.0f + erff(val * 0.70710678f));
                    outH[(size_t)gm * N + gn] = f2bf(val);
                } else if (EPI == 1) {
                    outF[(size_t)gm * N + gn] = val;
                    outH[(size_t)gm * N + gn] = f2bf(val);
                } else {
                    if (mask[gm >> 2] == 0) val = 0.f;   // row -> (b,t) = row/DD
                    outF[(size_t)gm * N + gn] = val;
                }
            }
        }
    }
}

// ---------------------------------------------------------------------------
// Fused embed-add + LayerNorm, write bf16 S (one block per (b,t,d) row)
// ---------------------------------------------------------------------------
__global__ __launch_bounds__(256) void prep_ln_kernel(
    const float* __restrict__ H, const int* __restrict__ ids,
    const float* __restrict__ emb, const float* __restrict__ demb,
    const float* __restrict__ g, const float* __restrict__ be,
    unsigned short* __restrict__ Sln)
{
    const int row = blockIdx.x;
    const int bt = row >> 2, d = row & 3;
    const int tok = ids[bt];
    const int tid = threadIdx.x;

    float x[4], s = 0.f, ss = 0.f;
#pragma unroll
    for (int j = 0; j < 4; ++j) {
        int   i = j * 256 + tid;
        float v = H[(size_t)bt * DMc + i] + emb[(size_t)tok * DMc + i] + demb[d * DMc + i];
        x[j] = v; s += v; ss += v * v;
    }
    __shared__ float rs[256], rq[256];
    rs[tid] = s; rq[tid] = ss;
    __syncthreads();
    for (int off = 128; off > 0; off >>= 1) {
        if (tid < off) { rs[tid] += rs[tid + off]; rq[tid] += rq[tid + off]; }
        __syncthreads();
    }
    const float mu   = rs[0] * (1.f / DMc);
    const float var  = rq[0] * (1.f / DMc) - mu * mu;
    const float rstd = rsqrtf(var + EPSc);
#pragma unroll
    for (int j = 0; j < 4; ++j) {
        int   i = j * 256 + tid;
        float v = (x[j] - mu) * rstd * g[i] + be[i];
        Sln[(size_t)row * DMc + i] = f2bf(v);
    }
}

// ---------------------------------------------------------------------------
// beta = S2 @ Wb + bb (masked) ; k = dynamic branch width
// ---------------------------------------------------------------------------
__global__ __launch_bounds__(256) void beta_k_kernel(
    const float* __restrict__ S2f, const float* __restrict__ Wb,
    const float* __restrict__ bbias, const int* __restrict__ mask,
    float* __restrict__ betaOut, int* __restrict__ kOut)
{
    const int row = blockIdx.x;
    const int tid = threadIdx.x;
    float p = 0.f;
    for (int i = tid; i < DMc; i += 256) p += S2f[(size_t)row * DMc + i] * Wb[i];
    __shared__ float rs[256];
    rs[tid] = p;
    __syncthreads();
    for (int off = 128; off > 0; off >>= 1) {
        if (tid < off) rs[tid] += rs[tid + off];
        __syncthreads();
    }
    if (tid == 0) {
        const int m    = mask[row >> 2];
        float     beta = m ? (rs[0] + bbias[0]) : 0.f;
        betaOut[row] = beta;
        float sig = 1.f / (1.f + expf(-beta));
        int   k   = (int)ceilf(sig * (float)Kbr);
        k = k < 1 ? 1 : (k > Kbr ? Kbr : k);
        if (sig < 0.25f) k = 0;
        if (!m) k = 0;
        kOut[row] = k;
    }
}

// ---------------------------------------------------------------------------
// Masked token CE: per-row two-pass logsumexp over V, atomic accumulate
// ---------------------------------------------------------------------------
__global__ __launch_bounds__(256) void ce_kernel(
    const float* __restrict__ q, const int* __restrict__ mask,
    const int* __restrict__ labels, float* __restrict__ accum)
{
    const int row = blockIdx.x;
    const int bt = row >> 2, d = row & 3;
    const int b = bt / Tc, t = bt % Tc;
    const int shift = d + 1;
    const bool valid = (t + shift < Tc) && mask[bt] && mask[b * Tc + t + shift];
    if (!valid) return;

    const int    tgt = labels[b * Tc + t + shift];
    const float* qr  = q + (size_t)row * Vc;
    const int    tid = threadIdx.x;

    float mx = -1e30f;
    for (int i = tid; i < Vc; i += 256) mx = fmaxf(mx, qr[i]);
    __shared__ float rs[256];
    rs[tid] = mx;
    __syncthreads();
    for (int off = 128; off > 0; off >>= 1) {
        if (tid < off) rs[tid] = fmaxf(rs[tid], rs[tid + off]);
        __syncthreads();
    }
    const float rowmax = rs[0];
    __syncthreads();

    float se = 0.f;
    for (int i = tid; i < Vc; i += 256) se += expf(qr[i] - rowmax);
    rs[tid] = se;
    __syncthreads();
    for (int off = 128; off > 0; off >>= 1) {
        if (tid < off) rs[tid] += rs[tid + off];
        __syncthreads();
    }
    if (tid == 0) {
        float nll = logf(rs[0]) + rowmax - qr[tgt];
        atomicAdd(&accum[0], nll);
        atomicAdd(&accum[1], 1.f);
    }
}

// ---------------------------------------------------------------------------
// BCE-with-logits on beta over source positions with any valid depth
// ---------------------------------------------------------------------------
__global__ void bce_kernel(const float* __restrict__ beta, const int* __restrict__ mask,
                           float* __restrict__ accum)
{
    int bt = blockIdx.x * blockDim.x + threadIdx.x;
    if (bt >= Bc * Tc) return;
    int  b = bt / Tc, t = bt % Tc;
    bool vd[4], src = false;
#pragma unroll
    for (int d = 0; d < 4; ++d) {
        int  shift = d + 1;
        bool v = (t + shift < Tc) && mask[bt] && mask[b * Tc + t + shift];
        vd[d] = v; src = src || v;
    }
    if (!src) return;
    float s = 0.f;
#pragma unroll
    for (int d = 0; d < 4; ++d) {
        float x  = beta[bt * 4 + d];
        float sp = fmaxf(x, 0.f) + log1pf(expf(-fabsf(x)));  // logaddexp(0, x)
        s += sp - x * (vd[d] ? 1.f : 0.f);
    }
    atomicAdd(&accum[2], s);
    atomicAdd(&accum[3], 1.f);
}

__global__ void init_kernel(float* accum) {
    if (threadIdx.x < 4) accum[threadIdx.x] = 0.f;
}

__global__ void finalize_kernel(const float* __restrict__ accum, float* __restrict__ lossOut) {
    float cnt = accum[1];
    float Lq  = cnt > 0.f ? accum[0] / fmaxf(cnt, 1.f) : 0.f;
    float Lb  = accum[2] / fmaxf(accum[3], 1.f);
    *lossOut = Lq + Lb;
}

// ---------------------------------------------------------------------------
extern "C" void kernel_launch(void* const* d_in, const int* in_sizes, int n_in,
                              void* d_out, int out_size, void* d_ws, size_t ws_size,
                              hipStream_t stream)
{
    const float* H      = (const float*)d_in[0];
    const int*   ids    = (const int*)d_in[1];
    const int*   amask  = (const int*)d_in[2];
    const int*   labels = (const int*)d_in[3];
    const float* emb    = (const float*)d_in[4];
    const float* demb   = (const float*)d_in[5];
    const float* ln_g   = (const float*)d_in[6];
    const float* ln_b   = (const float*)d_in[7];
    const float* W1     = (const float*)d_in[8];
    const float* b1     = (const float*)d_in[9];
    const float* W2     = (const float*)d_in[10];
    const float* b2     = (const float*)d_in[11];
    const float* Wq     = (const float*)d_in[12];
    const float* bq     = (const float*)d_in[13];
    const float* Wb     = (const float*)d_in[14];
    const float* bb     = (const float*)d_in[15];

    const int M = Bc * Tc * DDc;  // 4096 draft rows

    char*  ws  = (char*)d_ws;
    size_t off = 256;
    float*          accum = (float*)ws;
    unsigned short* Sln = (unsigned short*)(ws + off); off += (size_t)M * DMc * 2;
    unsigned short* U   = (unsigned short*)(ws + off); off += (size_t)M * DHc * 2;
    float*          S2f = (float*)(ws + off);          off += (size_t)M * DMc * 4;
    unsigned short* S2h = (unsigned short*)(ws + off); off += (size_t)M * DMc * 2;
    unsigned short* Wt1 = (unsigned short*)(ws + off); off += (size_t)DMc * DHc * 2;
    unsigned short* Wt2 = (unsigned short*)(ws + off); off += (size_t)DHc * DMc * 2;
    unsigned short* Wtq = (unsigned short*)(ws + off); off += (size_t)DMc * Vc * 2;

    float* q       = (float*)d_out;                 // (B,T,DD,V)
    float* betaOut = q + (size_t)M * Vc;            // (B,T,DD)
    int*   kOut    = (int*)(betaOut + M);           // (B,T,DD) int32
    float* lossOut = (float*)(kOut + M);            // scalar

    init_kernel<<<1, 32, 0, stream>>>(accum);

    // one-time bf16 transposed weight copies (amortized over 32 M-passes each)
    convert_transpose_kernel<<<dim3(DHc / 32, DMc / 32), 256, 0, stream>>>(W1, Wt1, DMc, DHc);
    convert_transpose_kernel<<<dim3(DMc / 32, DHc / 32), 256, 0, stream>>>(W2, Wt2, DHc, DMc);
    convert_transpose_kernel<<<dim3(Vc / 32, DMc / 32), 256, 0, stream>>>(Wq, Wtq, DMc, Vc);

    prep_ln_kernel<<<M, 256, 0, stream>>>(H, ids, emb, demb, ln_g, ln_b, Sln);

    gemm_bf16_wmma<0><<<dim3(DHc / 128, M / 128), 256, 0, stream>>>(
        Sln, Wt1, b1, M, DHc, DMc, U, nullptr, nullptr);
    gemm_bf16_wmma<1><<<dim3(DMc / 128, M / 128), 256, 0, stream>>>(
        U, Wt2, b2, M, DMc, DHc, S2h, S2f, nullptr);
    gemm_bf16_wmma<2><<<dim3(Vc / 128, M / 128), 256, 0, stream>>>(
        S2h, Wtq, bq, M, Vc, DMc, nullptr, q, amask);

    beta_k_kernel<<<M, 256, 0, stream>>>(S2f, Wb, bb, amask, betaOut, kOut);
    ce_kernel<<<M, 256, 0, stream>>>(q, amask, labels, accum);
    bce_kernel<<<(Bc * Tc + 255) / 256, 256, 0, stream>>>(betaOut, amask, accum);
    finalize_kernel<<<1, 1, 0, stream>>>(accum, lossOut);
}
